// DbrxExperts_49228915147015
// MI455X (gfx1250) — matrix operand order repaired and checked by reading, not compiled, not run on hardware
//
#include <hip/hip_runtime.h>
#include <hip/hip_bf16.h>
#include <math.h>

#define T_TOK 2048
#define HDIM  1024
#define FDIM  2048
#define NEXP  8
#define TOPK  2

typedef __attribute__((ext_vector_type(16))) __bf16         v16bf;
typedef __attribute__((ext_vector_type(8)))  float          v8f;
typedef __attribute__((ext_vector_type(8)))  unsigned short ushort8;

union Frag {
  v16bf   bf;
  ushort8 u8[2];
};

// fp32 -> bf16 round-to-nearest-even (used by the convert kernels and the GEMM1 epilogue)
static __device__ __forceinline__ unsigned short f2bf(float f) {
  union { float f; unsigned u; } v; v.f = f;
  unsigned r = v.u + 0x7FFFu + ((v.u >> 16) & 1u);
  return (unsigned short)(r >> 16);
}

// CDNA5 async global->LDS copy, 16B per lane, tracked by ASYNCcnt.
static __device__ __forceinline__ void async_copy16(unsigned lds_off, const void* gptr) {
  asm volatile("global_load_async_to_lds_b128 %0, %1, off"
               :: "v"(lds_off), "v"((unsigned long long)(size_t)gptr)
               : "memory");
}
static __device__ __forceinline__ void wait_async() {
  asm volatile("s_wait_asynccnt 0" ::: "memory");
}

__global__ void zero_out_kernel(float* __restrict__ out, int n) {
  int i = blockIdx.x * blockDim.x + threadIdx.x;
  if (i < n) out[i] = 0.0f;
}

// cw[t][e] = sum over top-k slots routed to e of top_weights[t,k]
__global__ void router_kernel(const float* __restrict__ tw,
                              const int* __restrict__ te,
                              float* __restrict__ cw) {
  int t = blockIdx.x * blockDim.x + threadIdx.x;
  if (t >= T_TOK) return;
  float c[NEXP];
#pragma unroll
  for (int e = 0; e < NEXP; ++e) c[e] = 0.0f;
#pragma unroll
  for (int k = 0; k < TOPK; ++k) {
    int e = te[t * TOPK + k] & (NEXP - 1);
    c[e] += tw[t * TOPK + k];
  }
#pragma unroll
  for (int e = 0; e < NEXP; ++e) cw[t * NEXP + e] = c[e];
}

// One-shot fp32 -> bf16, 8 elements/thread (2x float4 load, 1x b128 store).
__global__ void convert_bf16_kernel(const float* __restrict__ src,
                                    unsigned short* __restrict__ dst, int n8) {
  int i = blockIdx.x * blockDim.x + threadIdx.x;
  if (i >= n8) return;
  const float4* s = (const float4*)src + (size_t)i * 2;
  float4 a = s[0];
  float4 b = s[1];
  ushort8 o;
  o[0] = f2bf(a.x); o[1] = f2bf(a.y); o[2] = f2bf(a.z); o[3] = f2bf(a.w);
  o[4] = f2bf(b.x); o[5] = f2bf(b.y); o[6] = f2bf(b.z); o[7] = f2bf(b.w);
  *((ushort8*)dst + i) = o;
}

// One-shot fp32 [E][F][H] -> bf16 [E][H][F] (LDS-tiled transpose, coalesced both sides).
__global__ __launch_bounds__(256)
void transpose_convert_kernel(const float* __restrict__ src,
                              unsigned short* __restrict__ dst) {
  __shared__ unsigned short tile[32][33];
  const int e = blockIdx.z;
  const float* s = src + (size_t)e * FDIM * HDIM;
  unsigned short* d = dst + (size_t)e * FDIM * HDIM;
  const int h0 = blockIdx.x * 32;
  const int f0 = blockIdx.y * 32;
  const int tx = threadIdx.x & 31;
  const int ty = threadIdx.x >> 5;   // 0..7
#pragma unroll
  for (int j = 0; j < 4; ++j)
    tile[ty + j * 8][tx] = f2bf(s[(size_t)(f0 + ty + j * 8) * HDIM + h0 + tx]);
  __syncthreads();
#pragma unroll
  for (int j = 0; j < 4; ++j)
    d[(size_t)(h0 + ty + j * 8) * FDIM + f0 + tx] = tile[tx][ty + j * 8];
}

// GEMM1: gate = x @ w1e^T, up = x @ v1e^T, inter = silu(gate)*up (bf16 out).
// All operands pre-converted bf16; w1t/v1t pre-transposed to [H][F] so the
// B tile ([k][n], n contiguous) is a straight async byte copy into LDS.
// Block: 64(t) x 64(f); waves 0-3 gate / 4-7 up; 4 WMMA accumulators per wave.
__global__ __launch_bounds__(256)
void moe_gemm1_kernel(const unsigned short* __restrict__ xb,    // [T][H]
                      const unsigned short* __restrict__ w1t,   // [H][F]
                      const unsigned short* __restrict__ v1t,   // [H][F]
                      unsigned short* __restrict__ inter) {     // [T][F]
  // 4 B-tiles (2 buffers x 2 matrices), each [32][72] bf16 (padded, 16B-aligned rows)
  __shared__ __align__(16) unsigned char smem[18432];
  typedef unsigned short BTile[32][72];
  BTile* Bs = (BTile*)smem;          // Bs[buf*2 + mat]
  float* xch = (float*)smem;         // reused after the k-loop (16KB < 18432)

  const int tid  = threadIdx.x;
  const int wave = tid >> 5;
  const int lane = tid & 31;
  const int f0 = blockIdx.x * 64;
  const int t0 = blockIdx.y * 64;
  const bool gatew = (wave < 4);
  const int msub = (wave & 3) * 16;
  const int lr = lane & 15;
  const int kb = (lane >> 4) * 8;

  // staging role: one 16B segment per thread per matrix per k-slab
  const int srow = tid >> 3;         // 0..31 : k within slab
  const int sseg = (tid & 7) * 8;    // n offset (halves)
  const unsigned short* gw = w1t + (size_t)srow * FDIM + f0 + sseg;
  const unsigned short* gv = v1t + (size_t)srow * FDIM + f0 + sseg;
  const unsigned short* arow = xb + (size_t)(t0 + msub + lr) * HDIM;

  v8f acc[4];
#pragma unroll
  for (int nf = 0; nf < 4; ++nf)
#pragma unroll
    for (int r = 0; r < 8; ++r) acc[nf][r] = 0.0f;

  // prologue: stage slab 0 into buffer 0
  async_copy16((unsigned)(size_t)&Bs[0][srow][sseg], gw);
  async_copy16((unsigned)(size_t)&Bs[1][srow][sseg], gv);
  wait_async();
  __syncthreads();

  for (int k0 = 0; k0 < HDIM; k0 += 32) {
    const int cur = (k0 >> 5) & 1;
    if (k0 + 32 < HDIM) {            // overlap next slab's async copy with WMMA
      const size_t goff = (size_t)(k0 + 32) * FDIM;
      async_copy16((unsigned)(size_t)&Bs[(cur ^ 1) * 2 + 0][srow][sseg], gw + goff);
      async_copy16((unsigned)(size_t)&Bs[(cur ^ 1) * 2 + 1][srow][sseg], gv + goff);
    }
    Frag a;                          // A fragment straight from global bf16 (L2-hot)
    a.u8[0] = *(const ushort8*)(arow + k0 + kb);
    a.u8[1] = *(const ushort8*)(arow + k0 + kb + 16);
    const unsigned short (*B)[72] = Bs[cur * 2 + (gatew ? 0 : 1)];
#pragma unroll
    for (int nf = 0; nf < 4; ++nf) {
      Frag b;                        // lane%16 = n, halves = k (table order)
#pragma unroll
      for (int s = 0; s < 8; ++s) {
        b.u8[0][s] = B[kb + s][nf * 16 + lr];
        b.u8[1][s] = B[16 + kb + s][nf * 16 + lr];
      }
      acc[nf] = __builtin_amdgcn_wmma_f32_16x16x32_bf16(
          false, a.bf, false, b.bf, (short)0, acc[nf], false, false);
    }
    wait_async();
    __syncthreads();
  }

  // Exchange: up waves (4-7) hand accumulators to gate partner (w-4) via LDS.
  if (!gatew) {
    float* dst = xch + (wave - 4) * 1024;
#pragma unroll
    for (int nf = 0; nf < 4; ++nf)
#pragma unroll
      for (int r = 0; r < 8; ++r)
        dst[nf * 256 + r * 32 + lane] = acc[nf][r];
  }
  __syncthreads();
  if (gatew) {
    const float* src = xch + wave * 1024;
    const int mhi = (lane & 16) ? 8 : 0;
#pragma unroll
    for (int nf = 0; nf < 4; ++nf) {
#pragma unroll
      for (int r = 0; r < 8; ++r) {
        float g = acc[nf][r];
        float u = src[nf * 256 + r * 32 + lane];
        float s = g / (1.0f + __expf(-g));   // silu
        int m = msub + r + mhi;
        int n = nf * 16 + (lane & 15);
        inter[(size_t)(t0 + m) * FDIM + f0 + n] = f2bf(s * u);
      }
    }
  }
}

// GEMM2: out[t,h] += cw[t][e] * sum_f inter[t,f] * w2e[f,h]  (w2e pre-converted bf16)
__global__ __launch_bounds__(256)
void moe_gemm2_kernel(const unsigned short* __restrict__ inter,  // [T][F]
                      const unsigned short* __restrict__ w2b,    // [F][H]
                      const float* __restrict__ cw,
                      int eIdx,
                      float* __restrict__ out) {
  __shared__ __align__(16) unsigned short Bs[2][32][72];  // double-buffered [k][n]

  const int tid  = threadIdx.x;
  const int wave = tid >> 5;
  const int lane = tid & 31;
  const int h0 = blockIdx.x * 64;
  const int t0 = blockIdx.y * 64;
  const int msub  = (wave & 3) * 16;
  const int nbase = (wave >> 2) * 32;
  const int lr = lane & 15;
  const int kb = (lane >> 4) * 8;

  const int srow = tid >> 3;
  const int sseg = (tid & 7) * 8;
  const unsigned short* gw = w2b + (size_t)srow * HDIM + h0 + sseg;
  const unsigned short* arow = inter + (size_t)(t0 + msub + lr) * FDIM;

  v8f acc[2];
#pragma unroll
  for (int nf = 0; nf < 2; ++nf)
#pragma unroll
    for (int r = 0; r < 8; ++r) acc[nf][r] = 0.0f;

  async_copy16((unsigned)(size_t)&Bs[0][srow][sseg], gw);
  wait_async();
  __syncthreads();

  for (int k0 = 0; k0 < FDIM; k0 += 32) {
    const int cur = (k0 >> 5) & 1;
    if (k0 + 32 < FDIM)
      async_copy16((unsigned)(size_t)&Bs[cur ^ 1][srow][sseg],
                   gw + (size_t)(k0 + 32) * HDIM);
    Frag a;
    a.u8[0] = *(const ushort8*)(arow + k0 + kb);
    a.u8[1] = *(const ushort8*)(arow + k0 + kb + 16);
#pragma unroll
    for (int nf = 0; nf < 2; ++nf) {
      Frag b;
#pragma unroll
      for (int s = 0; s < 8; ++s) {
        b.u8[0][s] = Bs[cur][kb + s][nbase + nf * 16 + lr];
        b.u8[1][s] = Bs[cur][16 + kb + s][nbase + nf * 16 + lr];
      }
      acc[nf] = __builtin_amdgcn_wmma_f32_16x16x32_bf16(
          false, a.bf, false, b.bf, (short)0, acc[nf], false, false);
    }
    wait_async();
    __syncthreads();
  }

  // out += cw * acc; each (t,h) owned by one block, experts stream-serialized.
  const int mhi = (lane & 16) ? 8 : 0;
#pragma unroll
  for (int r = 0; r < 8; ++r) {
    int t = t0 + msub + r + mhi;
    float c = cw[(size_t)t * NEXP + eIdx];
#pragma unroll
    for (int nf = 0; nf < 2; ++nf) {
      int h = h0 + nbase + nf * 16 + (lane & 15);
      out[(size_t)t * HDIM + h] += c * acc[nf][r];
    }
  }
}

extern "C" void kernel_launch(void* const* d_in, const int* in_sizes, int n_in,
                              void* d_out, int out_size, void* d_ws, size_t ws_size,
                              hipStream_t stream) {
  const float* x  = (const float*)d_in[0];
  // d_in[1] = full router probs: unused (as in reference)
  const float* tw = (const float*)d_in[2];
  const int*   te = (const int*)d_in[3];
  const float* w1 = (const float*)d_in[4];
  const float* v1 = (const float*)d_in[5];
  const float* w2 = (const float*)d_in[6];
  float* out = (float*)d_out;

  // Workspace layout (bytes):
  //   cw   [T][E] fp32 :      0 ..  64K
  //   inter[T][F] bf16 :    64K ..  8M+64K
  //   xb   [T][H] bf16 :  +8M   (4 MB)
  //   w1t  [E][H][F]   :  +4M   (32 MB)   (transposed)
  //   v1t  [E][H][F]   :  +32M  (32 MB)   (transposed)
  //   w2b  [E][F][H]   :  +32M  (32 MB)
  char* ws = (char*)d_ws;
  float*          cw    = (float*)ws;
  unsigned short* inter = (unsigned short*)(ws + 65536);
  unsigned short* xb    = (unsigned short*)(ws + 65536 + (size_t)8  * 1024 * 1024);
  unsigned short* w1t   = (unsigned short*)(ws + 65536 + (size_t)12 * 1024 * 1024);
  unsigned short* v1t   = (unsigned short*)(ws + 65536 + (size_t)44 * 1024 * 1024);
  unsigned short* w2b   = (unsigned short*)(ws + 65536 + (size_t)76 * 1024 * 1024);

  zero_out_kernel<<<(T_TOK * HDIM + 255) / 256, 256, 0, stream>>>(out, T_TOK * HDIM);
  router_kernel<<<(T_TOK + 255) / 256, 256, 0, stream>>>(tw, te, cw);

  // One-shot precision conversion (bandwidth-bound, done once per launch).
  convert_bf16_kernel<<<(T_TOK * HDIM / 8 + 255) / 256, 256, 0, stream>>>(
      x, xb, T_TOK * HDIM / 8);
  convert_bf16_kernel<<<(NEXP * FDIM * HDIM / 8 + 255) / 256, 256, 0, stream>>>(
      w2, w2b, NEXP * FDIM * HDIM / 8);
  transpose_convert_kernel<<<dim3(HDIM / 32, FDIM / 32, NEXP), 256, 0, stream>>>(w1, w1t);
  transpose_convert_kernel<<<dim3(HDIM / 32, FDIM / 32, NEXP), 256, 0, stream>>>(v1, v1t);

  const size_t esz = (size_t)FDIM * HDIM;
  for (int e = 0; e < NEXP; ++e) {
    moe_gemm1_kernel<<<dim3(FDIM / 64, T_TOK / 64), 256, 0, stream>>>(
        xb, w1t + (size_t)e * esz, v1t + (size_t)e * esz, inter);
    moe_gemm2_kernel<<<dim3(HDIM / 64, T_TOK / 64), 256, 0, stream>>>(
        inter, w2b + (size_t)e * esz, cw, e, out);
  }
}
